// RNN_MultiRegional_D1D2_87617332838687
// MI455X (gfx1250) — matrix operand order repaired and checked by reading, not compiled, not run on hardware
//
#include <hip/hip_runtime.h>

// ---------------- problem constants ----------------
#define HID      512
#define FSI      153
#define UNITS_   3225          // 6*HID + FSI
#define UPAD     3264          // padded: 64*51 = 16*204
#define BATCH    64
#define TSTEPS   500
#define KSTEP    64            // K advanced per LDS stage (2 WMMAs per wave)
#define KTILES   51            // UPAD / KSTEP
#define NTILES   204           // UPAD / 16
#define NBLOCKS  102           // 32 output columns per block
#define ALM_OFF  2713          // 5*HID + FSI
#define ALM_EXC  359
#define LDSS     72            // LDS row stride in bf16 (144B: 16B-aligned, bank-skewed)
#define BUFU     (96 * LDSS)   // ushorts per LDS buffer: 64 A rows + 32 B rows
#define NOISE_SC 0.0044721359549995794f   // 0.01 * sqrt(2*0.1)

typedef __attribute__((ext_vector_type(16))) __bf16 v16bf;
typedef __attribute__((ext_vector_type(8)))  float  v8f;
typedef __attribute__((ext_vector_type(4)))  int    v4i;

union FragB16 { uint4 q[2]; v16bf v; };

#if __has_builtin(__builtin_amdgcn_global_load_async_to_lds_b128)
#define USE_ASYNC 1
typedef __attribute__((address_space(1))) v4i* glb_i4p;
typedef __attribute__((address_space(3))) v4i* lds_i4p;
#else
#define USE_ASYNC 0
#endif

static __device__ __forceinline__ void wait_async0() {
#if USE_ASYNC
#if __has_builtin(__builtin_amdgcn_s_wait_asynccnt)
  __builtin_amdgcn_s_wait_asynccnt(0);
#else
  asm volatile("s_wait_asynccnt 0" ::: "memory");
#endif
#endif
}

static __device__ __forceinline__ unsigned short f2bf(float f) {
  unsigned int u = __float_as_uint(f);
  unsigned int r = u + 0x7FFFu + ((u >> 16) & 1u);   // round-to-nearest-even
  return (unsigned short)(r >> 16);
}

static __device__ __forceinline__ float ht(float w) {   // hardtanh(w, 1e-10, 1)
  return fminf(fmaxf(w, 1e-10f), 1.0f);
}

static __device__ __forceinline__ int region(int u, int* loc) {
  if (u < 512)  { *loc = u;        return 0; }  // str
  if (u < 665)  { *loc = u - 512;  return 1; }  // fsi
  if (u < 1177) { *loc = u - 665;  return 2; }  // gpe
  if (u < 1689) { *loc = u - 1177; return 3; }  // stn
  if (u < 2201) { *loc = u - 1689; return 4; }  // snr
  if (u < 2713) { *loc = u - 2201; return 5; }  // thal
  *loc = u - 2713; return 6;                    // alm
}

// ------------- build W (bf16, [dst n][src k], zero padded to UPAD x UPAD) -------------
__global__ __launch_bounds__(256) void build_w_kernel(
    const float* __restrict__ s2s,  const float* __restrict__ t2a,
    const float* __restrict__ t2s,  const float* __restrict__ a2a,
    const float* __restrict__ a2s,  const float* __restrict__ a2t,
    const float* __restrict__ s2snr,const float* __restrict__ s2gpe,
    const float* __restrict__ g2stn,const float* __restrict__ stn2snr,
    const float* __restrict__ snr2t,const float* __restrict__ f2s,
    const float* __restrict__ t2f,  const float* __restrict__ a2f,
    const float* __restrict__ f2f,  const int* __restrict__ smask,
    unsigned short* __restrict__ Wb)
{
  int idx = blockIdx.x * blockDim.x + threadIdx.x;
  if (idx >= UPAD * UPAD) return;
  int n = idx / UPAD, k = idx % UPAD;
  float v = 0.f;
  if (n < UNITS_ && k < UNITS_) {
    int nl, kl;
    int rn = region(n, &nl), rk = region(k, &kl);
    if (rn == 0) {                                            // -> str
      if (rk == 0) v = -((smask[nl*512+kl] ? 1.f : 0.f) * ht(s2s[nl*512+kl]));
      else if (rk == 1) v = -ht(f2s[nl*153+kl]);
      else if (rk == 5) v = (nl < 256) ? ht(t2s[nl*512+kl]) : 0.f;   // D1 rows
      else if (rk == 6) v = (kl < ALM_EXC) ? ht(a2s[nl*512+kl]) : 0.f;
    } else if (rn == 1) {                                     // -> fsi
      if (rk == 1) v = -ht(f2f[nl*153+kl]);
      else if (rk == 5) v = ht(t2f[nl*512+kl]);
      else if (rk == 6) v = (kl < ALM_EXC) ? ht(a2f[nl*512+kl]) : 0.f;
    } else if (rn == 2) {                                     // -> gpe (D2 cols)
      if (rk == 0) v = (kl >= 256) ? -ht(s2gpe[nl*512+kl]) : 0.f;
    } else if (rn == 3) {                                     // -> stn
      if (rk == 2) v = -ht(g2stn[nl*512+kl]);
    } else if (rn == 4) {                                     // -> snr (D1 cols)
      if (rk == 0) v = (kl < 256) ? -ht(s2snr[nl*512+kl]) : 0.f;
      else if (rk == 3) v = ht(stn2snr[nl*512+kl]);
    } else if (rn == 5) {                                     // -> thal
      if (rk == 4) v = -ht(snr2t[nl*512+kl]);
      else if (rk == 6) v = (kl < ALM_EXC) ? ht(a2t[nl*512+kl]) : 0.f;
    } else {                                                  // -> alm
      if (rk == 5) v = ht(t2a[nl*512+kl]);
      else if (rk == 6) v = ht(a2a[nl*512+kl]) * ((kl < ALM_EXC) ? 1.f : -1.f);
    }
  }
  Wb[idx] = f2bf(v);
}

// ------------- init: per-unit vectors, x0, h0 (bf16 ping-pong) -------------
__global__ __launch_bounds__(256) void init_kernel(
    const float* __restrict__ xn, const float* __restrict__ hn,
    const float* __restrict__ inp_w_in,
    float* __restrict__ x, unsigned short* __restrict__ hbf0,
    unsigned short* __restrict__ hbf1,
    float* __restrict__ tonic, float* __restrict__ inpw,
    float* __restrict__ thalm)
{
  int idx = blockIdx.x * blockDim.x + threadIdx.x;
  if (idx < UPAD) {
    int u = idx;
    float tn = 0.f;
    if (u < UNITS_)
      tn = (u < 665) ? 0.1f : (u < 1177) ? 0.6f : (u < 1689) ? 0.3f
                                         : (u < 2201) ? 0.6f : 0.1f;
    tonic[u] = tn;
    inpw[u]  = (u < HID + FSI) ? inp_w_in[u] : 0.f;
    thalm[u] = (u >= 2201 && u < 2713) ? 1.f : 0.f;
  } else if (idx < UPAD + BATCH * UPAD) {
    int j = idx - UPAD;
    int b = j / UPAD, u = j % UPAD;
    float xv = (u < UNITS_) ? xn[b * UNITS_ + u] : 0.f;
    float hv = (u < UNITS_) ? hn[b * UNITS_ + u] : 0.f;
    x[(size_t)b * UPAD + u]    = xv;
    hbf0[(size_t)b * UPAD + u] = f2bf(hv);
    hbf1[(size_t)b * UPAD + u] = f2bf(hv);
  }
}

// ------------- one RNN step: drive = h @ W^T (+ elementwise), integrate, relu -------------
// 256 threads = 8 waves: wave w -> (M tile = w&3, N tile = w>>2); block covers 32 columns.
// Double-buffered LDS stage of A(64xKSTEP) + B(32xKSTEP); one barrier per stage.
__global__ __launch_bounds__(256) void step_kernel(
    const unsigned short* __restrict__ Wb,
    const unsigned short* __restrict__ hprev,
    unsigned short* __restrict__ hnext,
    float* __restrict__ x,
    const float* __restrict__ tonic, const float* __restrict__ inpw,
    const float* __restrict__ thalm,
    const float* __restrict__ inp,   const float* __restrict__ cue,
    const float* __restrict__ inhib, const float* __restrict__ noiseh,
    const float* __restrict__ noisei,
    float* __restrict__ hs, int t)
{
  __shared__ unsigned short smem[2 * BUFU];   // [buf][64 A rows | 32 B rows] x LDSS
  __shared__ float ldsIE[BATCH];
  __shared__ float ldsCue[BATCH];

  const int tid   = threadIdx.x;
  const int lane  = tid & 31;          // wave32
  const int wv    = tid >> 5;          // 0..7
  const int mtile = wv & 3;            // M tile (batch rows)
  const int ntile = wv >> 2;           // N tile within block (0..1)
  const int half  = lane >> 4;
  const int l16   = lane & 15;
  const int nbase = blockIdx.x * 32;

  if (tid < BATCH) {
    int b = tid;
    ldsIE[b]  = inp[b * TSTEPS + t] + NOISE_SC * noisei[t * BATCH + b];
    ldsCue[b] = cue[b * TSTEPS + t];
  }

  // cooperative stage assignments: all 256 threads participate
  const int arow = tid >> 2;            // 0..63 (4 threads per A row)
  const int acol = (tid & 3) * 16;      // 16 bf16 = 32B per thread (2 x b128)
  const int brow = tid >> 3;            // 0..31 (8 threads per B row)
  const int bcol = (tid & 7) * 8;       // 8 bf16 = 16B per thread (1 x b128)

  const unsigned short* pAg = hprev + (size_t)arow * UPAD + acol;
  const unsigned short* pBg = Wb + (size_t)(nbase + brow) * UPAD + bcol;
  const int sAoff = arow * LDSS + acol;              // within buffer
  const int sBoff = 64 * LDSS + brow * LDSS + bcol;

#if USE_ASYNC
  // stage kt=0 into buffer 0 via async global->LDS (ASYNCcnt-tracked)
  __builtin_amdgcn_global_load_async_to_lds_b128(
      (glb_i4p)(pAg), (lds_i4p)(smem + sAoff), 0, 0);
  __builtin_amdgcn_global_load_async_to_lds_b128(
      (glb_i4p)(pAg + 8), (lds_i4p)(smem + sAoff + 8), 0, 0);
  __builtin_amdgcn_global_load_async_to_lds_b128(
      (glb_i4p)(pBg), (lds_i4p)(smem + sBoff), 0, 0);
#else
  {
    uint4 a0 = *(const uint4*)(pAg);
    uint4 a1 = *(const uint4*)(pAg + 8);
    uint4 b0 = *(const uint4*)(pBg);
    *(uint4*)(smem + sAoff)     = a0;
    *(uint4*)(smem + sAoff + 8) = a1;
    *(uint4*)(smem + sBoff)     = b0;
  }
#endif

  v8f acc = {};

  for (int kt = 0; kt < KTILES; ++kt) {
    const int p = kt & 1;
    unsigned short* buf  = smem + p * BUFU;
    unsigned short* nbuf = smem + (p ^ 1) * BUFU;

#if USE_ASYNC
    wait_async0();          // my async writes into buf p done
    __syncthreads();        // everyone's writes done; buf p^1 fully consumed
    if (kt + 1 < KTILES) {
      const int kn = (kt + 1) * KSTEP;
      __builtin_amdgcn_global_load_async_to_lds_b128(
          (glb_i4p)(pAg + kn), (lds_i4p)(nbuf + sAoff), 0, 0);
      __builtin_amdgcn_global_load_async_to_lds_b128(
          (glb_i4p)(pAg + kn + 8), (lds_i4p)(nbuf + sAoff + 8), 0, 0);
      __builtin_amdgcn_global_load_async_to_lds_b128(
          (glb_i4p)(pBg + kn), (lds_i4p)(nbuf + sBoff), 0, 0);
    }
#else
    __syncthreads();        // buf p published; buf p^1 fully consumed
    uint4 a0, a1, b0;
    if (kt + 1 < KTILES) {
      const int kn = (kt + 1) * KSTEP;
      a0 = *(const uint4*)(pAg + kn);
      a1 = *(const uint4*)(pAg + kn + 8);
      b0 = *(const uint4*)(pBg + kn);
    }
#endif

    #pragma unroll
    for (int s = 0; s < 2; ++s) {       // two 16x16x32 sub-steps
      FragB16 fa, fb;
      // A fragment: row M = l16 of this wave's M tile (ISA 16-bit A layout)
      const unsigned short* pa = buf + (mtile * 16 + l16) * LDSS + s * 32 + half * 8;
      fa.q[0] = *(const uint4*)(pa);        // K = half*8 + 0..7
      fa.q[1] = *(const uint4*)(pa + 16);   // K = half*8 + 16..23
      // B fragment: col N = l16 of this wave's N tile, K = half*8 + 0..7, half*8 + 16..23
      const unsigned short* pb = buf + 64 * LDSS + (ntile * 16 + l16) * LDSS + s * 32 + half * 8;
      fb.q[0] = *(const uint4*)(pb);
      fb.q[1] = *(const uint4*)(pb + 16);

      acc = __builtin_amdgcn_wmma_f32_16x16x32_bf16(
          false, fa.v, false, fb.v, (short)0, acc, false, false);
    }

#if !USE_ASYNC
    if (kt + 1 < KTILES) {
      *(uint4*)(nbuf + sAoff)     = a0;
      *(uint4*)(nbuf + sAoff + 8) = a1;
      *(uint4*)(nbuf + sBoff)     = b0;
    }
#endif
  }

  // epilogue: C/D layout -> element (M = mtile*16 + half*8 + r, N = nbase + ntile*16 + l16)
  const int u = nbase + ntile * 16 + l16;
  if (u < UNITS_) {
    const float tn = tonic[u], iw = inpw[u], tm = thalm[u];
    #pragma unroll
    for (int r = 0; r < 8; ++r) {
      const int m = mtile * 16 + half * 8 + r;     // batch index
      float drive = acc[r]
                  + ldsIE[m] * iw
                  + ldsCue[m] * tm
                  + tn
                  + inhib[((size_t)m * TSTEPS + t) * UNITS_ + u]
                  + NOISE_SC * noiseh[((size_t)t * BATCH + m) * UNITS_ + u];
      const size_t xi = (size_t)m * UPAD + u;
      float xo  = x[xi];
      float xnw = 0.9f * xo + 0.1f * drive;        // x + 0.1*(-x + drive)
      float hv  = fmaxf(xnw, 0.f);
      x[xi] = xnw;
      hs[((size_t)m * TSTEPS + t) * UNITS_ + u] = hv;
      hnext[xi] = f2bf(hv);
    }
  }
}

// ------------- readout: out[b,t] = h[b,t, alm_exc] . out_w -------------
__global__ __launch_bounds__(64) void out_kernel(
    const float* __restrict__ hs, const float* __restrict__ ow,
    float* __restrict__ outs, int t)
{
  __shared__ float s[64];
  const int b = blockIdx.x, tid = threadIdx.x;
  float p = 0.f;
  for (int j = tid; j < ALM_EXC; j += 64)
    p += hs[((size_t)b * TSTEPS + t) * UNITS_ + ALM_OFF + j] * ow[j];
  s[tid] = p;
  __syncthreads();
  if (tid == 0) {
    float sum = 0.f;
    #pragma unroll
    for (int i = 0; i < 64; ++i) sum += s[i];
    outs[b * TSTEPS + t] = sum;
  }
}

static inline size_t align256(size_t x) { return (x + 255) & ~(size_t)255; }

extern "C" void kernel_launch(void* const* d_in, const int* in_sizes, int n_in,
                              void* d_out, int out_size, void* d_ws, size_t ws_size,
                              hipStream_t stream) {
  const float* inp    = (const float*)d_in[0];
  const float* cue    = (const float*)d_in[1];
  const float* inhib  = (const float*)d_in[2];
  const float* hn     = (const float*)d_in[3];
  const float* xn     = (const float*)d_in[4];
  const float* noiseh = (const float*)d_in[5];
  const float* noisei = (const float*)d_in[6];
  const float* s2s    = (const float*)d_in[7];
  const float* t2a    = (const float*)d_in[8];
  const float* t2s    = (const float*)d_in[9];
  const float* a2a    = (const float*)d_in[10];
  const float* a2s    = (const float*)d_in[11];
  const float* a2t    = (const float*)d_in[12];
  const float* s2snr  = (const float*)d_in[13];
  const float* s2gpe  = (const float*)d_in[14];
  const float* g2stn  = (const float*)d_in[15];
  const float* stn2snr= (const float*)d_in[16];
  const float* snr2t  = (const float*)d_in[17];
  const float* f2s    = (const float*)d_in[18];
  const float* t2f    = (const float*)d_in[19];
  const float* a2f    = (const float*)d_in[20];
  const float* f2f    = (const float*)d_in[21];
  const float* inp_w  = (const float*)d_in[22];
  const float* out_w  = (const float*)d_in[23];
  const int*   smask  = (const int*)d_in[24];

  float* hs   = (float*)d_out;                               // [B, T, UNITS]
  float* outs = hs + (size_t)BATCH * TSTEPS * UNITS_;        // [B, T, 1]

  char* ws = (char*)d_ws;
  size_t off = 0;
  unsigned short* Wb   = (unsigned short*)(ws + off); off = align256(off + (size_t)UPAD * UPAD * 2);
  unsigned short* hbf0 = (unsigned short*)(ws + off); off = align256(off + (size_t)BATCH * UPAD * 2);
  unsigned short* hbf1 = (unsigned short*)(ws + off); off = align256(off + (size_t)BATCH * UPAD * 2);
  float* xbuf  = (float*)(ws + off); off = align256(off + (size_t)BATCH * UPAD * 4);
  float* tonic = (float*)(ws + off); off = align256(off + (size_t)UPAD * 4);
  float* inpwv = (float*)(ws + off); off = align256(off + (size_t)UPAD * 4);
  float* thalm = (float*)(ws + off); off = align256(off + (size_t)UPAD * 4);
  (void)ws_size; (void)in_sizes; (void)n_in; (void)out_size;

  {
    const int total = UPAD * UPAD;
    build_w_kernel<<<(total + 255) / 256, 256, 0, stream>>>(
        s2s, t2a, t2s, a2a, a2s, a2t, s2snr, s2gpe, g2stn, stn2snr, snr2t,
        f2s, t2f, a2f, f2f, smask, Wb);
  }
  {
    const int total = UPAD + BATCH * UPAD;
    init_kernel<<<(total + 255) / 256, 256, 0, stream>>>(
        xn, hn, inp_w, xbuf, hbf0, hbf1, tonic, inpwv, thalm);
  }

  for (int t = 0; t < TSTEPS; ++t) {
    const unsigned short* hp = (t & 1) ? hbf1 : hbf0;
    unsigned short*       hx = (t & 1) ? hbf0 : hbf1;
    step_kernel<<<NBLOCKS, 256, 0, stream>>>(
        Wb, hp, hx, xbuf, tonic, inpwv, thalm,
        inp, cue, inhib, noiseh, noisei, hs, t);
    out_kernel<<<BATCH, 64, 0, stream>>>(hs, out_w, outs, t);
  }
}
